// MoELayer_18519898980909
// MI455X (gfx1250) — matrix operand order repaired and checked by reading, not compiled, not run on hardware
//
#include <hip/hip_runtime.h>
#include <hip/hip_bf16.h>
#include <stdint.h>

// Problem constants: B=2, N=2048 -> T=4096, C=1024, F=4096, E=8, K=2
#define T_TOK 4096
#define C_DIM 1024
#define F_DIM 4096
#define E_NUM 8
#define MTILE 32                  // tokens per block (2 WMMA M-tiles)
#define TILES (T_TOK / MTILE)     // 128 row-tiles per expert (worst case)
#define FCHUNK 256                // F processed per fused chunk
#define XS_P (C_DIM + 8)          // padded LDS row: 2064B stride (+4 bank rotation/row)
#define HS_P (FCHUNK + 8)

typedef __bf16 v16bf __attribute__((ext_vector_type(16)));
typedef float  v8f   __attribute__((ext_vector_type(8)));
typedef unsigned int tdm_v4u __attribute__((ext_vector_type(4)));
typedef int          tdm_v8i __attribute__((ext_vector_type(8)));
typedef int          tdm_v4i __attribute__((ext_vector_type(4)));

union BF16x16 {
    v16bf v;
    uint4 q[2];
    unsigned short us[16];
};

__device__ __forceinline__ unsigned short f2bf(float f) {
    unsigned int u = __float_as_uint(f);
    u += 0x7FFFu + ((u >> 16) & 1u);   // round-to-nearest-even
    return (unsigned short)(u >> 16);
}

// ---------------------------------------------------------------------------
// Kernel 1: convert w1 and w2 (fp32) to bf16 in workspace, vectorized x4.
// ---------------------------------------------------------------------------
__global__ __launch_bounds__(256) void moe_convert_weights(
    const float4* __restrict__ w1, const float4* __restrict__ w2,
    uint2* __restrict__ w1b, uint2* __restrict__ w2b, int n4) {
    int tid = blockIdx.x * 256 + threadIdx.x;
    const float4* src;
    uint2* dst;
    int i;
    if (tid < n4) { src = w1; dst = w1b; i = tid; }
    else          { src = w2; dst = w2b; i = tid - n4; }
    float4 f = src[i];
    uint2 o;
    o.x = (unsigned)f2bf(f.x) | ((unsigned)f2bf(f.y) << 16);
    o.y = (unsigned)f2bf(f.z) | ((unsigned)f2bf(f.w) << 16);
    dst[i] = o;
}

// ---------------------------------------------------------------------------
// Kernel 2: router. One wave32 per token. logits -> softmax -> top-2 -> renorm,
// atomic-append token to per-expert lists; converts x row to bf16.
// ---------------------------------------------------------------------------
__global__ __launch_bounds__(256) void moe_router(
    const float* __restrict__ x, const float* __restrict__ rw,
    unsigned short* __restrict__ xb, unsigned int* __restrict__ counts,
    int* __restrict__ tok_idx, float* __restrict__ tok_w) {
    __shared__ float rwl[E_NUM * C_DIM];   // 32 KB
    for (int i = threadIdx.x; i < E_NUM * C_DIM / 4; i += 256)
        ((float4*)rwl)[i] = ((const float4*)rw)[i];
    __syncthreads();

    int wave = threadIdx.x >> 5;
    int lane = threadIdx.x & 31;
    int t = blockIdx.x * 8 + wave;

    const float4* xr = (const float4*)(x + (size_t)t * C_DIM);
    uint2* xo = (uint2*)(xb + (size_t)t * C_DIM);

    float acc[E_NUM];
#pragma unroll
    for (int e = 0; e < E_NUM; ++e) acc[e] = 0.f;

#pragma unroll
    for (int j = 0; j < 8; ++j) {
        int idx = j * 32 + lane;
        float4 f = xr[idx];
        uint2 o;
        o.x = (unsigned)f2bf(f.x) | ((unsigned)f2bf(f.y) << 16);
        o.y = (unsigned)f2bf(f.z) | ((unsigned)f2bf(f.w) << 16);
        xo[idx] = o;
#pragma unroll
        for (int e = 0; e < E_NUM; ++e) {
            float4 r = ((const float4*)(rwl + e * C_DIM))[idx];
            acc[e] += f.x * r.x + f.y * r.y + f.z * r.z + f.w * r.w;
        }
    }
#pragma unroll
    for (int e = 0; e < E_NUM; ++e)
#pragma unroll
        for (int off = 16; off > 0; off >>= 1)
            acc[e] += __shfl_xor(acc[e], off, 32);

    if (lane == 0) {
        float m = acc[0];
#pragma unroll
        for (int e = 1; e < E_NUM; ++e) m = fmaxf(m, acc[e]);
        float p[E_NUM];
#pragma unroll
        for (int e = 0; e < E_NUM; ++e) p[e] = __expf(acc[e] - m);
        int i1 = 0;
#pragma unroll
        for (int e = 1; e < E_NUM; ++e) if (p[e] > p[i1]) i1 = e;
        int i2 = (i1 == 0) ? 1 : 0;
#pragma unroll
        for (int e = 0; e < E_NUM; ++e) if (e != i1 && p[e] > p[i2]) i2 = e;
        float denom = p[i1] + p[i2] + 1e-20f;
        float wa = p[i1] / denom;
        float wb = p[i2] / denom;
        unsigned pos1 = atomicAdd(&counts[i1], 1u);
        tok_idx[i1 * T_TOK + pos1] = t;
        tok_w[i1 * T_TOK + pos1] = wa;
        unsigned pos2 = atomicAdd(&counts[i2], 1u);
        tok_idx[i2 * T_TOK + pos2] = t;
        tok_w[i2 * T_TOK + pos2] = wb;
    }
}

// ---------------------------------------------------------------------------
// Kernel 3: fused grouped FFN. Block = (expert, 32-token tile), 16 waves.
// x-tile is gathered into LDS by the Tensor Data Mover (gather-mode D#, 16
// row indices per descriptor, LDS pad 8B/1KB -> 2064B row stride) when the
// TDM builtin exists; plain staging otherwise (identical LDS layout).
// Phase 1: h = relu(x @ w1^T + b1) per 256-wide F chunk, B double-buffered.
// Phase 2: y += h @ w2^T, 4 B fragments clause-loaded per K step, 8 WMMAs.
// Epilogue: out[t] += route_w * (y + b2) via fp32 global atomics.
// ---------------------------------------------------------------------------
__global__ __launch_bounds__(512) void moe_ffn(
    const unsigned short* __restrict__ xb,
    const unsigned short* __restrict__ w1b,
    const unsigned short* __restrict__ w2b,
    const float* __restrict__ b1, const float* __restrict__ b2,
    const unsigned int* __restrict__ counts,
    const int* __restrict__ tok_idx, const float* __restrict__ tok_w,
    float* __restrict__ out) {
    __shared__ unsigned short xs[MTILE][XS_P];   // ~64.5 KB token tile (bf16)
    __shared__ unsigned short hs[MTILE][HS_P];   // ~16.5 KB activation chunk
    __shared__ int   toks[MTILE];
    __shared__ float tws[MTILE];

    int e    = blockIdx.x / TILES;
    int tile = blockIdx.x % TILES;
    int rows = (int)counts[e];
    int row0 = tile * MTILE;
    if (row0 >= rows) return;                   // uniform across block
    int valid = min(MTILE, rows - row0);

    if (threadIdx.x < MTILE) {
        int r = threadIdx.x;
        if (r < valid) {
            toks[r] = tok_idx[e * T_TOK + row0 + r];
            tws[r]  = tok_w[e * T_TOK + row0 + r];
        } else {
            toks[r] = -1;
            tws[r]  = 0.f;
        }
    }
    __syncthreads();

    int wave = threadIdx.x >> 5;
    int lane = threadIdx.x & 31;
    int l16  = lane & 15;
    int hi   = lane >> 4;        // A/B K-half select, C/D row-half select

    // ---- Stage gathered x rows into LDS ----
    // LDS row layout (both paths): 2064B stride, 8B hole after byte 1024
    // (element offset: c*2 + (c>=512 ? 8 : 0)).
#if __has_builtin(__builtin_amdgcn_tensor_load_to_lds)
    if (wave == 0) {
        unsigned lds0 = (unsigned)(uintptr_t)&xs[0][0];
        unsigned long long ga = (unsigned long long)(uintptr_t)xb;
#pragma unroll
        for (int d = 0; d < 2; ++d) {
            tdm_v4u g0;
            g0[0] = 1u | (1u << 31);                    // count=1, gather_mode, 16b idx
            g0[1] = lds0 + d * 16 * (XS_P * 2);         // lds_addr
            g0[2] = (unsigned)(ga & 0xFFFFFFFFull);     // global_addr[31:0]
            g0[3] = (unsigned)((ga >> 32) & 0x1FFFFFFull) | (2u << 30);  // [56:32], type=2
            tdm_v8i g1;
            // data_size=8B, pad_enable, pad_interval=256 DW (1KB), pad_amount=2 DW (8B)
            g1[0] = (int)((3u << 16) | (1u << 20) | (7u << 22) | (1u << 25));
            g1[1] = (int)(256u << 16);                  // tensor_dim0 = 256 (8B units)
            g1[2] = (int)(4096u << 16);                 // tensor_dim1 = 4096 rows
            g1[3] = (int)(256u << 16);                  // tile_dim0 = 256 (full row)
            g1[4] = 16;                                 // tile_dim1 = #row indices
            g1[5] = 256;                                // tensor_dim0_stride (8B units)
            g1[6] = 0;
            g1[7] = 0;
            tdm_v4i g2, g3;
#pragma unroll
            for (int i = 0; i < 4; ++i) {
                int r0 = d * 16 + 2 * i;
                int ta = toks[r0];     ta = ta < 0 ? 0 : ta;
                int tb = toks[r0 + 1]; tb = tb < 0 ? 0 : tb;
                g2[i] = (ta & 0xFFFF) | (tb << 16);
                int r8 = r0 + 8;
                int tc_ = toks[r8];     tc_ = tc_ < 0 ? 0 : tc_;
                int td  = toks[r8 + 1]; td  = td  < 0 ? 0 : td;
                g3[i] = (tc_ & 0xFFFF) | (td << 16);
            }
            tdm_v8i g4 = {0, 0, 0, 0, 0, 0, 0, 0};      // unused extra SGPR group
            __builtin_amdgcn_tensor_load_to_lds(g0, g1, g2, g3, g4, 0);
        }
        __builtin_amdgcn_s_wait_tensorcnt(0);
    }
#else
    for (int i = threadIdx.x; i < MTILE * (C_DIM / 8); i += 512) {
        int r = i / (C_DIM / 8);
        int cpart = i % (C_DIM / 8);
        uint4 v = make_uint4(0u, 0u, 0u, 0u);
        int t = toks[r];
        if (t >= 0) v = ((const uint4*)(xb + (size_t)t * C_DIM))[cpart];
        *(uint4*)&xs[r][cpart * 8 + (cpart >= 64 ? 4 : 0)] = v;
    }
#endif

    v8f accY[2][4];
#pragma unroll
    for (int m = 0; m < 2; ++m)
#pragma unroll
        for (int tc = 0; tc < 4; ++tc)
            accY[m][tc] = (v8f){0.f,0.f,0.f,0.f,0.f,0.f,0.f,0.f};

    size_t w1base = (size_t)e * F_DIM * C_DIM;
    size_t w2base = (size_t)e * C_DIM * F_DIM;
    int fl0   = wave * 16;       // this wave's 16 f-columns within chunk
    int cbase = wave * 64;       // this wave's 64 C-columns

    for (int fc = 0; fc < F_DIM / FCHUNK; ++fc) {
        int fc0 = fc * FCHUNK;
        __syncthreads();         // x/TDM ready (fc==0) / hs consumed (fc>0)

        // ---- Phase 1: h[32 x 16] per wave, K over C=1024, B double-buffered ----
        v8f accH[2];
        accH[0] = (v8f){0.f,0.f,0.f,0.f,0.f,0.f,0.f,0.f};
        accH[1] = (v8f){0.f,0.f,0.f,0.f,0.f,0.f,0.f,0.f};
        int frow = fc0 + fl0 + l16;
        const unsigned short* bp1 = w1b + w1base + (size_t)frow * C_DIM + hi * 8;
        BF16x16 Bcur;
        Bcur.q[0] = *(const uint4*)(bp1);
        Bcur.q[1] = *(const uint4*)(bp1 + 16);
        for (int kc = 0; kc < C_DIM / 32 - 1; ++kc) {
            int kb = kc * 32;
            int kbx = kb + (kb >= 512 ? 4 : 0);        // skip 8B TDM pad hole
            BF16x16 A0, A1, Bnxt;
            A0.q[0] = *(const uint4*)&xs[l16][kbx + hi * 8];
            A0.q[1] = *(const uint4*)&xs[l16][kbx + 16 + hi * 8];
            A1.q[0] = *(const uint4*)&xs[16 + l16][kbx + hi * 8];
            A1.q[1] = *(const uint4*)&xs[16 + l16][kbx + 16 + hi * 8];
            Bnxt.q[0] = *(const uint4*)(bp1 + kb + 32);
            Bnxt.q[1] = *(const uint4*)(bp1 + kb + 48);
            __builtin_prefetch(bp1 + kb + 64, 0, 3);
            accH[0] = __builtin_amdgcn_wmma_f32_16x16x32_bf16(
                false, A0.v, false, Bcur.v, (short)0, accH[0], false, false);
            accH[1] = __builtin_amdgcn_wmma_f32_16x16x32_bf16(
                false, A1.v, false, Bcur.v, (short)0, accH[1], false, false);
            Bcur = Bnxt;
        }
        {   // peeled last K step (kb = 992)
            int kbx = 992 + 4;
            BF16x16 A0, A1;
            A0.q[0] = *(const uint4*)&xs[l16][kbx + hi * 8];
            A0.q[1] = *(const uint4*)&xs[l16][kbx + 16 + hi * 8];
            A1.q[0] = *(const uint4*)&xs[16 + l16][kbx + hi * 8];
            A1.q[1] = *(const uint4*)&xs[16 + l16][kbx + 16 + hi * 8];
            accH[0] = __builtin_amdgcn_wmma_f32_16x16x32_bf16(
                false, A0.v, false, Bcur.v, (short)0, accH[0], false, false);
            accH[1] = __builtin_amdgcn_wmma_f32_16x16x32_bf16(
                false, A1.v, false, Bcur.v, (short)0, accH[1], false, false);
        }
        // bias + relu, write bf16 h chunk to LDS
        {
            int fcol = fc0 + fl0 + l16;
            float bv = b1[(size_t)e * F_DIM + fcol];
#pragma unroll
            for (int m = 0; m < 2; ++m)
#pragma unroll
                for (int j = 0; j < 8; ++j) {
                    float v = accH[m][j] + bv;       // C/D: row = j + 8*hi, col = l16
                    v = v > 0.f ? v : 0.f;
                    hs[m * 16 + j + hi * 8][fl0 + l16] = f2bf(v);
                }
        }
        __syncthreads();

        // ---- Phase 2: y[32 x 64] per wave; 4 B fragments per K step ----
        for (int kc = 0; kc < FCHUNK / 32; ++kc) {
            int kb = kc * 32;
            BF16x16 A0, A1, Bv[4];
            A0.q[0] = *(const uint4*)&hs[l16][kb + hi * 8];
            A0.q[1] = *(const uint4*)&hs[l16][kb + 16 + hi * 8];
            A1.q[0] = *(const uint4*)&hs[16 + l16][kb + hi * 8];
            A1.q[1] = *(const uint4*)&hs[16 + l16][kb + 16 + hi * 8];
#pragma unroll
            for (int tc = 0; tc < 4; ++tc) {
                int ccol = cbase + tc * 16 + l16;
                const unsigned short* bp =
                    w2b + w2base + (size_t)ccol * F_DIM + fc0 + kb + hi * 8;
                Bv[tc].q[0] = *(const uint4*)(bp);
                Bv[tc].q[1] = *(const uint4*)(bp + 16);
                __builtin_prefetch(bp + 32, 0, 3);
            }
#pragma unroll
            for (int tc = 0; tc < 4; ++tc) {
                accY[0][tc] = __builtin_amdgcn_wmma_f32_16x16x32_bf16(
                    false, A0.v, false, Bv[tc].v, (short)0, accY[0][tc], false, false);
                accY[1][tc] = __builtin_amdgcn_wmma_f32_16x16x32_bf16(
                    false, A1.v, false, Bv[tc].v, (short)0, accY[1][tc], false, false);
            }
        }
    }

    // ---- Epilogue: out[t, c] += route_w * (y + b2[c]) ----
#pragma unroll
    for (int tc = 0; tc < 4; ++tc) {
        int ccol = cbase + tc * 16 + l16;
        float b2v = b2[(size_t)e * C_DIM + ccol];
#pragma unroll
        for (int m = 0; m < 2; ++m)
#pragma unroll
            for (int j = 0; j < 8; ++j) {
                int r = m * 16 + j + hi * 8;
                if (r < valid) {
                    int t = toks[r];
                    float wt = tws[r];
                    unsafeAtomicAdd(&out[(size_t)t * C_DIM + ccol],
                                    wt * (accY[m][tc][j] + b2v));
                }
            }
    }
}

// ---------------------------------------------------------------------------
extern "C" void kernel_launch(void* const* d_in, const int* in_sizes, int n_in,
                              void* d_out, int out_size, void* d_ws, size_t ws_size,
                              hipStream_t stream) {
    const float* x  = (const float*)d_in[0];
    const float* rw = (const float*)d_in[1];
    const float* w1 = (const float*)d_in[2];
    const float* b1 = (const float*)d_in[3];
    const float* w2 = (const float*)d_in[4];
    const float* b2 = (const float*)d_in[5];
    float* out = (float*)d_out;

    char* ws = (char*)d_ws;
    size_t off = 0;
    unsigned short* xb  = (unsigned short*)(ws + off); off += (size_t)T_TOK * C_DIM * 2;
    unsigned short* w1b = (unsigned short*)(ws + off); off += (size_t)E_NUM * F_DIM * C_DIM * 2;
    unsigned short* w2b = (unsigned short*)(ws + off); off += (size_t)E_NUM * C_DIM * F_DIM * 2;
    unsigned int* counts = (unsigned int*)(ws + off);  off += 256;
    int*   tok_idx = (int*)(ws + off);                 off += (size_t)E_NUM * T_TOK * 4;
    float* tok_w   = (float*)(ws + off);               off += (size_t)E_NUM * T_TOK * 4;

    (void)hipMemsetAsync(counts, 0, 256, stream);
    (void)hipMemsetAsync(d_out, 0, (size_t)out_size * sizeof(float), stream);

    int n4 = E_NUM * F_DIM * C_DIM / 4;   // float4 count per weight tensor
    moe_convert_weights<<<2 * n4 / 256, 256, 0, stream>>>(
        (const float4*)w1, (const float4*)w2, (uint2*)w1b, (uint2*)w2b, n4);

    moe_router<<<T_TOK / 8, 256, 0, stream>>>(x, rw, xb, counts, tok_idx, tok_w);

    moe_ffn<<<E_NUM * TILES, 512, 0, stream>>>(
        xb, w1b, w2b, b1, b2, counts, tok_idx, tok_w, out);
}
